// CIN_65429531787416
// MI455X (gfx1250) — matrix operand order repaired and checked by reading, not compiled, not run on hardware
//
#include <hip/hip_runtime.h>

// ---------------- CDNA5 WMMA types ----------------
typedef __bf16 bf16;
typedef __attribute__((ext_vector_type(16))) __bf16 v16bf;
typedef __attribute__((ext_vector_type(8)))  float  v8f;

union BFrag {
    struct { uint4 lo, hi; } u;   // 8 dwords = 16 bf16
    v16bf v;
};

// Re-indexed K dimension:  k' = g * Fhp + f   (Fhp = Fh padded to mult. of 32)
//   -> every aligned 32-wide K-tile has constant g and contiguous f.
// Zero-padded f rows contribute Z=0, so the padded W slots are written as 0.

// ------------------------------------------------------------------
// Pre-pack W (C x 256, f32, rows k = f*40+g) into WMMA B-fragment order
// under the k' indexing, as bf16.
// B-matrix 32x16 (16-bit) layout: lane L holds column N = L%16;
// element e holds K = (e<8 ? e : e+8) + 8*(L>>4).
// One wave per (kt, nt) tile; dst[((kt*16+nt)*32+lane)] = 16 bf16 (32 B).
// ------------------------------------------------------------------
__global__ void cin_prepack_w(const float* __restrict__ W, uint4* __restrict__ dst,
                              int fshift, int Fh)
{
    const int tile = blockIdx.x;          // kt*16 + nt
    const int kt   = tile >> 4;
    const int nt   = tile & 15;
    const int lane = threadIdx.x;         // 0..31
    const int half = lane >> 4;
    const int n    = nt * 16 + (lane & 15);
    const int fmask = (1 << fshift) - 1;

    BFrag fr;
#pragma unroll
    for (int e = 0; e < 16; ++e) {
        const int kp = kt * 32 + (e < 8 ? e : e + 8) + 8 * half;   // k'
        const int g  = kp >> fshift;
        const int f  = kp & fmask;
        const float v = (f < Fh) ? W[((size_t)f * 40 + g) * 256 + n] : 0.f;
        fr.v[e] = (bf16)v;
    }
    uint4* p = dst + ((size_t)tile * 32 + lane) * 2;
    p[0] = fr.u.lo;
    p[1] = fr.u.hi;
}

// ------------------------------------------------------------------
// One CIN layer, one batch element per workgroup (4 waves, 128 threads).
//   Y(32 x 256) = Z(32 x C') @ W'(C' x 256) + bias, ReLU
//   Z[d, k'=g*Fhp+f] = h[f,d] * x0[g,d]
// Wave w owns N in [w*64, w*64+64): 2 M-tiles x 4 N-tiles (8 accums),
// so each A-fragment build feeds 8 WMMAs. K-loop unrolled 2x with
// double-buffered B fragments.
// ------------------------------------------------------------------
#define HT_STRIDE 132   // Fhp(max 128) + 4: 16B-aligned rows, de-conflicted banks

__global__ void __launch_bounds__(128)
cin_layer(const float* __restrict__ x0,     // (B, 40, 32)
          const float* __restrict__ h,      // (B, Fh, 32)
          int Fh,                            // 40 or 128
          int fshift,                        // log2(Fhp): 6 or 7
          int ktiles,                        // 40*Fhp/32 (80 or 160), always even
          const uint4* __restrict__ Wp,      // pre-packed weights
          const float* __restrict__ bias,    // (256,)
          float* __restrict__ out,           // (B, 512)
          int jbase,                         // 0 / 128 / 256
          float* __restrict__ hnext)         // (B, 128, 32) or nullptr
{
    __shared__ float x0_s[40 * 32];
    __shared__ float hT[32 * HT_STRIDE];     // transposed: hT[d*HT_STRIDE + f]

    const int b    = blockIdx.x;
    const int tid  = threadIdx.x;
    const int lane = tid & 31;
    const int wave = tid >> 5;               // 0..3
    const int half = lane >> 4;
    const int lm   = lane & 15;
    const int Fhp  = 1 << fshift;
    const int fmask = Fhp - 1;

    // ---- stage x0[b] (d contiguous) and h[b] transposed into LDS ----
    const float* x0g = x0 + (size_t)b * 40 * 32;
    for (int i = tid; i < 40 * 32; i += 128) x0_s[i] = x0g[i];

    const float* hg = h + (size_t)b * Fh * 32;
    for (int i = tid; i < Fh * 32; i += 128) {
        const int f = i >> 5, d = i & 31;
        hT[d * HT_STRIDE + f] = hg[i];
    }
    const int padc = (Fhp - Fh) * 32;        // zero-pad rows f in [Fh, Fhp)
    for (int i = tid; i < padc; i += 128) {
        const int f = Fh + (i >> 5), d = i & 31;
        hT[d * HT_STRIDE + f] = 0.f;
    }
    __syncthreads();

    v8f acc0[4] = {}, acc1[4] = {};          // [N-tile t] for M-tile 0 / 1

    const int d0 = lm;                       // A layout: lane%16 = M (= d)
    const int d1 = lm + 16;
    const float* hrow0 = hT + d0 * HT_STRIDE;
    const float* hrow1 = hT + d1 * HT_STRIDE;

    // A-fragment builder. A 16-bit 16x32 layout:
    //   K(e,half) = (e<8 ? e : e+8) + 8*half  ->  f-quads are contiguous.
    auto buildA = [&](BFrag& a0, BFrag& a1, int kt) {
        const int k0 = kt * 32;
        const int g  = k0 >> fshift;                 // constant over the tile
        const int fb = (k0 & fmask) + 8 * half;
        const float xv0 = x0_s[g * 32 + d0];
        const float xv1 = x0_s[g * 32 + d1];
        const float4 p00 = *(const float4*)(hrow0 + fb);
        const float4 p01 = *(const float4*)(hrow0 + fb + 4);
        const float4 p02 = *(const float4*)(hrow0 + fb + 16);
        const float4 p03 = *(const float4*)(hrow0 + fb + 20);
        const float4 p10 = *(const float4*)(hrow1 + fb);
        const float4 p11 = *(const float4*)(hrow1 + fb + 4);
        const float4 p12 = *(const float4*)(hrow1 + fb + 16);
        const float4 p13 = *(const float4*)(hrow1 + fb + 20);
        a0.v[0]=(bf16)(p00.x*xv0); a0.v[1]=(bf16)(p00.y*xv0); a0.v[2]=(bf16)(p00.z*xv0); a0.v[3]=(bf16)(p00.w*xv0);
        a0.v[4]=(bf16)(p01.x*xv0); a0.v[5]=(bf16)(p01.y*xv0); a0.v[6]=(bf16)(p01.z*xv0); a0.v[7]=(bf16)(p01.w*xv0);
        a0.v[8]=(bf16)(p02.x*xv0); a0.v[9]=(bf16)(p02.y*xv0); a0.v[10]=(bf16)(p02.z*xv0); a0.v[11]=(bf16)(p02.w*xv0);
        a0.v[12]=(bf16)(p03.x*xv0); a0.v[13]=(bf16)(p03.y*xv0); a0.v[14]=(bf16)(p03.z*xv0); a0.v[15]=(bf16)(p03.w*xv0);
        a1.v[0]=(bf16)(p10.x*xv1); a1.v[1]=(bf16)(p10.y*xv1); a1.v[2]=(bf16)(p10.z*xv1); a1.v[3]=(bf16)(p10.w*xv1);
        a1.v[4]=(bf16)(p11.x*xv1); a1.v[5]=(bf16)(p11.y*xv1); a1.v[6]=(bf16)(p11.z*xv1); a1.v[7]=(bf16)(p11.w*xv1);
        a1.v[8]=(bf16)(p12.x*xv1); a1.v[9]=(bf16)(p12.y*xv1); a1.v[10]=(bf16)(p12.z*xv1); a1.v[11]=(bf16)(p12.w*xv1);
        a1.v[12]=(bf16)(p13.x*xv1); a1.v[13]=(bf16)(p13.y*xv1); a1.v[14]=(bf16)(p13.z*xv1); a1.v[15]=(bf16)(p13.w*xv1);
    };

    // Packed-W pointer for this wave's four N-tiles (nt = wave*4 + t).
    const size_t kstep = (size_t)16 * 32 * 2;        // uint4 per K-tile
    const uint4* pw = Wp + (((size_t)wave * 4) * 32 + lane) * 2;
    auto loadB = [&](BFrag* f, const uint4* p) {
#pragma unroll
        for (int t = 0; t < 4; ++t) {
            f[t].u.lo = p[t * 64];
            f[t].u.hi = p[t * 64 + 1];
        }
    };

    auto wmma8 = [&](const BFrag& a0, const BFrag& a1, const BFrag* bb) {
#pragma unroll
        for (int t = 0; t < 4; ++t) {
            acc0[t] = __builtin_amdgcn_wmma_f32_16x16x32_bf16(false, a0.v, false, bb[t].v,
                                                              (short)0, acc0[t], false, false);
            acc1[t] = __builtin_amdgcn_wmma_f32_16x16x32_bf16(false, a1.v, false, bb[t].v,
                                                              (short)0, acc1[t], false, false);
        }
    };

    // Software-pipelined K loop (ktiles even): double-buffered B fragments.
    BFrag be[4], bo[4], a0, a1;
    loadB(be, pw);
    for (int kt = 0; kt < ktiles; kt += 2) {
        buildA(a0, a1, kt);
        loadB(bo, pw + kstep);                       // prefetch kt+1
        wmma8(a0, a1, be);

        buildA(a0, a1, kt + 1);
        if (kt + 2 < ktiles) loadB(be, pw + 2 * kstep);   // prefetch kt+2
        wmma8(a0, a1, bo);

        pw += 2 * kstep;
    }

    // ---- epilogue: bias + ReLU, pooled d-sum, next-layer h ----
    // C/D 16x16 f32 layout: lane L -> N = L%16, M = r + 8*(L>>4).
#pragma unroll
    for (int t = 0; t < 4; ++t) {
        const int n = wave * 64 + t * 16 + lm;
        const float bv = bias[n];
        const v8f am0 = acc0[t];
        const v8f am1 = acc1[t];

        float y0[8], y1[8];
        float ps = 0.f;
#pragma unroll
        for (int r = 0; r < 8; ++r) {
            float v0 = am0[r] + bv; v0 = v0 > 0.f ? v0 : 0.f;   // d = r+8*half
            float v1 = am1[r] + bv; v1 = v1 > 0.f ? v1 : 0.f;   // d = 16+r+8*half
            y0[r] = v0; y1[r] = v1;
            ps += v0 + v1;
        }
        ps += __shfl_xor(ps, 16);                    // combine the two M-halves

        if (hnext != nullptr && n >= 128) {
            float* hn = hnext + ((size_t)b * 128 + (n - 128)) * 32;
#pragma unroll
            for (int r = 0; r < 8; ++r) {
                hn[r + 8 * half]      = y0[r];
                hn[16 + r + 8 * half] = y1[r];
            }
        }
        const bool writeOut = (hnext == nullptr) || (n < 128);
        if (writeOut && half == 0)
            out[(size_t)b * 512 + jbase + n] = ps;
    }
}

// ------------------------------------------------------------------
extern "C" void kernel_launch(void* const* d_in, const int* in_sizes, int n_in,
                              void* d_out, int out_size, void* d_ws, size_t ws_size,
                              hipStream_t stream)
{
    const float* inputs = (const float*)d_in[0];   // (512, 40, 32)
    const float* W1 = (const float*)d_in[1];       // (1600, 256)
    const float* b1 = (const float*)d_in[2];
    const float* W2 = (const float*)d_in[3];       // (5120, 256)
    const float* b2 = (const float*)d_in[4];
    const float* W3 = (const float*)d_in[5];       // (5120, 256)
    const float* b3 = (const float*)d_in[6];
    float* out = (float*)d_out;

    const int B = 512;
    const int KT1 = 40 * 64 / 32;    // 80  (Fh=40  -> Fhp=64,  fshift=6)
    const int KT2 = 40 * 128 / 32;   // 160 (Fh=128 -> Fhp=128, fshift=7)

    // Workspace layout (256B aligned)
    char* ws = (char*)d_ws;
    size_t off = 0;
    auto take = [&](size_t bytes) {
        void* p = ws + off;
        off = (off + bytes + 255) & ~(size_t)255;
        return p;
    };
    uint4* W1p = (uint4*)take((size_t)KT1 * 16 * 32 * 32);  // 1.3 MB
    uint4* W2p = (uint4*)take((size_t)KT2 * 16 * 32 * 32);  // 2.5 MB
    uint4* W3p = (uint4*)take((size_t)KT2 * 16 * 32 * 32);  // 2.5 MB
    float* h2  = (float*)take((size_t)B * 128 * 32 * 4);    // 8 MB
    float* h3  = (float*)take((size_t)B * 128 * 32 * 4);    // 8 MB

    // Pre-pack weights into B-fragment bf16 layout under k' indexing.
    cin_prepack_w<<<KT1 * 16, 32, 0, stream>>>(W1, W1p, 6, 40);
    cin_prepack_w<<<KT2 * 16, 32, 0, stream>>>(W2, W2p, 7, 128);
    cin_prepack_w<<<KT2 * 16, 32, 0, stream>>>(W3, W3p, 7, 128);

    // Layer 1: h = inputs; out rows 0..127, h2 = rows 128..255
    cin_layer<<<B, 128, 0, stream>>>(inputs, inputs, 40, 6, KT1, W1p, b1, out, 0, h2);
    // Layer 2: h = h2; out rows 128..255, h3 = rows 128..255
    cin_layer<<<B, 128, 0, stream>>>(inputs, h2, 128, 7, KT2, W2p, b2, out, 128, h3);
    // Layer 3: h = h3; out rows 256..511
    cin_layer<<<B, 128, 0, stream>>>(inputs, h3, 128, 7, KT2, W3p, b3, out, 256, nullptr);
}
